// TextEncoder_61744449847730
// MI455X (gfx1250) — matrix operand order repaired and checked by reading, not compiled, not run on hardware
//
#include <hip/hip_runtime.h>

#define VOCAB 2000
#define EMBED 8
#define HIDN 256
#define BATCH 256
#define SEQL 256
#define G4 1024          // 4*HIDN gate width
#define NBLK 32
#define TPB 256
#define LDS_STRIDE 68    // 64 + 4 pad to avoid bank conflicts

typedef __attribute__((ext_vector_type(16))) __bf16 v16bf;
typedef __attribute__((ext_vector_type(8)))  float  v8f;

union FragU { v16bf v; uint4 u[2]; };

__device__ __forceinline__ unsigned short f2bf(float f) {
  unsigned int u = __float_as_uint(f);
  u += 0x7FFFu + ((u >> 16) & 1u);          // round-to-nearest-even
  return (unsigned short)(u >> 16);
}

// A fragment: 16x32 bf16. Lane l: row M0 + (l&15); K elements
// {h*8 .. h*8+7} and {16+h*8 .. 16+h*8+23? no, +7} with h = l>>4 (ISA 7.12.2).
__device__ __forceinline__ v16bf load_a(const unsigned short* base, int ld,
                                        int M0, int K0, int lane) {
  const int half = lane >> 4, m = lane & 15;
  const unsigned short* p = base + (size_t)(M0 + m) * ld + K0 + half * 8;
  FragU f;
  f.u[0] = *(const uint4*)(p);        // K0+h*8 .. +7
  f.u[1] = *(const uint4*)(p + 16);   // K0+16+h*8 .. +7
  return f.v;
}

// B fragment: 32x16 bf16 of the pre-transposed weight (K-major). Lane l holds
// column N0+(l&15), K rows {h*8 .. h*8+7} and {16+h*8 .. 16+h*8+7}, h = l>>4.
__device__ __forceinline__ v16bf load_b(const unsigned short* base, int ld,
                                        int K0, int N0, int lane) {
  const int half = lane >> 4, n = lane & 15;
  const unsigned short* p = base + (size_t)(K0 + half * 8) * ld + N0 + n;
  union { v16bf v; unsigned short s[16]; } f;
#pragma unroll
  for (int i = 0; i < 8; ++i) {
    f.s[i]     = p[(size_t)i * ld];
    f.s[8 + i] = p[(size_t)(16 + i) * ld];
  }
  return f.v;
}

__device__ __forceinline__ v8f wmma_bf16(v16bf a, v16bf b, v8f c) {
  return __builtin_amdgcn_wmma_f32_16x16x32_bf16(false, a, false, b,
                                                 (short)0, c, false, false);
}

__device__ __forceinline__ float sigm(float x) { return 1.0f / (1.0f + __expf(-x)); }

struct Params {
  const int* tokens; const int* lengths;
  const float* emb;
  const float* w_ih0f; const float* w_hh0f; const float* b_ih0; const float* b_hh0;
  const float* w_ih1f; const float* w_hh1f; const float* b_ih1; const float* b_hh1;
  const float* ff_wf;  const float* ff_b;   const float* gamma; const float* beta;
  unsigned short *w_hh0t, *w_ih1t, *w_hh1t, *w_ih0t, *ff_wt;   // bf16, K-major, gate-interleaved
  unsigned short *h1b0, *h1b1, *h2b0, *h2b1, *xt, *lastb;      // bf16 state
  float *c1, *c2, *last, *y;
  int *cnt, *gen;
  float* out;
};

// Device-scope sense(generation) barrier over the persistent grid.
__device__ __forceinline__ void grid_barrier(int* cnt, int* gen, int nblk) {
  __syncthreads();
  if (threadIdx.x == 0) {
    int g = __hip_atomic_load(gen, __ATOMIC_ACQUIRE, __HIP_MEMORY_SCOPE_AGENT);
    int prev = __hip_atomic_fetch_add(cnt, 1, __ATOMIC_ACQ_REL, __HIP_MEMORY_SCOPE_AGENT);
    if (prev == nblk - 1) {
      __hip_atomic_store(cnt, 0, __ATOMIC_RELAXED, __HIP_MEMORY_SCOPE_AGENT);
      __hip_atomic_store(gen, g + 1, __ATOMIC_RELEASE, __HIP_MEMORY_SCOPE_AGENT);
    } else {
      while (__hip_atomic_load(gen, __ATOMIC_ACQUIRE, __HIP_MEMORY_SCOPE_AGENT) == g) {
        __builtin_amdgcn_s_sleep(1);
      }
    }
  }
  __syncthreads();
}

// acc[s] over 16x16 tiles s=0..3 covering N0..N0+63, K = Ksteps*32.
__device__ __forceinline__ void gemm_acc(v8f* acc, const unsigned short* A, int lda,
                                         const unsigned short* Bm, int ldb,
                                         int M0, int N0, int Ksteps, int lane) {
  for (int kk = 0; kk < Ksteps; ++kk) {
    v16bf a = load_a(A, lda, M0, kk * 32, lane);
#pragma unroll
    for (int s = 0; s < 4; ++s) {
      v16bf b = load_b(Bm, ldb, kk * 32, N0 + s * 16, lane);
      acc[s] = wmma_bf16(a, b, acc[s]);
    }
  }
}

__global__ void __launch_bounds__(TPB) init_kernel(Params p) {
  const int tid = blockIdx.x * blockDim.x + threadIdx.x;
  const int nth = gridDim.x * blockDim.x;
  if (tid == 0) { *p.cnt = 0; *p.gen = 0; }
  // Recurrent weights: Wt[k][4*j+g] = W[g*256+j][k]  (bf16, K-major, gate-interleaved)
  for (int i = tid; i < HIDN * G4; i += nth) {
    int k = i >> 10, c = i & 1023, j = c >> 2, g = c & 3;
    int src = (g * HIDN + j) * HIDN + k;
    p.w_hh0t[i] = f2bf(p.w_hh0f[src]);
    p.w_ih1t[i] = f2bf(p.w_ih1f[src]);
    p.w_hh1t[i] = f2bf(p.w_hh1f[src]);
  }
  // Input-projection weights padded K=8 -> 32
  for (int i = tid; i < 32 * G4; i += nth) {
    int e = i >> 10, c = i & 1023, j = c >> 2, g = c & 3;
    p.w_ih0t[i] = (e < EMBED) ? f2bf(p.w_ih0f[(g * HIDN + j) * EMBED + e])
                              : (unsigned short)0;
  }
  // FF weight transpose: ff_wt[k][n] = ff_w[n][k]
  for (int i = tid; i < HIDN * HIDN; i += nth) {
    int k = i >> 8, n = i & 255;
    p.ff_wt[i] = f2bf(p.ff_wf[n * HIDN + k]);
  }
  // Zero LSTM state
  for (int i = tid; i < BATCH * HIDN; i += nth) {
    p.h1b0[i] = 0; p.h1b1[i] = 0; p.h2b0[i] = 0; p.h2b1[i] = 0;
    p.c1[i] = 0.f; p.c2[i] = 0.f;
  }
  // Stage embedded inputs for t = 0 (padded to K=32)
  for (int i = tid; i < BATCH * 32; i += nth) {
    int bb = i >> 5, e = i & 31;
    float v = (e < EMBED) ? p.emb[p.tokens[bb * SEQL] * EMBED + e] : 0.f;
    p.xt[i] = f2bf(v);
  }
}

__global__ void __launch_bounds__(TPB) lstm_persistent(Params p) {
  __shared__ float lds_tiles[8 * 16 * LDS_STRIDE];
  const int lane = threadIdx.x & 31;
  const int wib  = threadIdx.x >> 5;
  const int wg   = blockIdx.x * 8 + wib;          // 0..255 global wave id
  const int M0   = (wg >> 4) * 16;                // batch-row tile
  const int J0   = (wg & 15) * 16;                // hidden-feature tile
  const int N0   = J0 * 4;                        // gate-interleaved column base
  float* tile = &lds_tiles[wib * 16 * LDS_STRIDE];
  const v8f zero8 = {0.f,0.f,0.f,0.f,0.f,0.f,0.f,0.f};
  const int gtid = blockIdx.x * TPB + threadIdx.x;  // 0..8191

  for (int t = 0; t < SEQL; ++t) {
    const unsigned short* h1r = (t & 1) ? p.h1b1 : p.h1b0;
    unsigned short*       h1w = (t & 1) ? p.h1b0 : p.h1b1;
    const unsigned short* h2r = (t & 1) ? p.h2b1 : p.h2b0;
    unsigned short*       h2w = (t & 1) ? p.h2b0 : p.h2b1;

    // ---------- phase 0: layer-0 gates + cell update ----------
    {
      v8f acc[4] = {zero8, zero8, zero8, zero8};
      gemm_acc(acc, h1r, HIDN, p.w_hh0t, G4, M0, N0, 8, lane);   // h1 @ Whh0'
      gemm_acc(acc, p.xt, 32, p.w_ih0t, G4, M0, N0, 1, lane);    // x  @ Wih0' (K padded)
#pragma unroll
      for (int r = 0; r < 8; ++r) {
        int row = r + 8 * (lane >> 4);
#pragma unroll
        for (int s = 0; s < 4; ++s)
          tile[row * LDS_STRIDE + s * 16 + (lane & 15)] = acc[s][r];
      }
      __syncthreads();
#pragma unroll
      for (int r = 0; r < 8; ++r) {
        int pq = lane + 32 * r;
        int m = pq >> 4, j = pq & 15;
        int bb = M0 + m, jj = J0 + j;
        const float* g4 = &tile[m * LDS_STRIDE + 4 * j];
        float gi = g4[0] + p.b_ih0[jj]            + p.b_hh0[jj];
        float gf = g4[1] + p.b_ih0[HIDN + jj]     + p.b_hh0[HIDN + jj];
        float gg = g4[2] + p.b_ih0[2 * HIDN + jj] + p.b_hh0[2 * HIDN + jj];
        float go = g4[3] + p.b_ih0[3 * HIDN + jj] + p.b_hh0[3 * HIDN + jj];
        float c = p.c1[bb * HIDN + jj];
        c = sigm(gf) * c + sigm(gi) * tanhf(gg);
        p.c1[bb * HIDN + jj] = c;
        h1w[bb * HIDN + jj] = f2bf(sigm(go) * tanhf(c));
      }
    }
    grid_barrier(p.cnt, p.gen, NBLK);

    // ---------- phase 1: layer-1 gates + cell update + last capture ----------
    {
      v8f acc[4] = {zero8, zero8, zero8, zero8};
      gemm_acc(acc, h1w, HIDN, p.w_ih1t, G4, M0, N0, 8, lane);   // new h1 @ Wih1'
      gemm_acc(acc, h2r, HIDN, p.w_hh1t, G4, M0, N0, 8, lane);   // old h2 @ Whh1'
#pragma unroll
      for (int r = 0; r < 8; ++r) {
        int row = r + 8 * (lane >> 4);
#pragma unroll
        for (int s = 0; s < 4; ++s)
          tile[row * LDS_STRIDE + s * 16 + (lane & 15)] = acc[s][r];
      }
      __syncthreads();
#pragma unroll
      for (int r = 0; r < 8; ++r) {
        int pq = lane + 32 * r;
        int m = pq >> 4, j = pq & 15;
        int bb = M0 + m, jj = J0 + j;
        const float* g4 = &tile[m * LDS_STRIDE + 4 * j];
        float gi = g4[0] + p.b_ih1[jj]            + p.b_hh1[jj];
        float gf = g4[1] + p.b_ih1[HIDN + jj]     + p.b_hh1[HIDN + jj];
        float gg = g4[2] + p.b_ih1[2 * HIDN + jj] + p.b_hh1[2 * HIDN + jj];
        float go = g4[3] + p.b_ih1[3 * HIDN + jj] + p.b_hh1[3 * HIDN + jj];
        float c = p.c2[bb * HIDN + jj];
        c = sigm(gf) * c + sigm(gi) * tanhf(gg);
        p.c2[bb * HIDN + jj] = c;
        float h2 = sigm(go) * tanhf(c);
        h2w[bb * HIDN + jj] = f2bf(h2);
        if (p.lengths[bb] - 1 == t) p.last[bb * HIDN + jj] = h2;  // ragged capture
      }
      // stage x for step t+1
      if (t + 1 < SEQL) {
        int bb = gtid >> 5, e = gtid & 31;
        float v = (e < EMBED) ? p.emb[p.tokens[bb * SEQL + t + 1] * EMBED + e] : 0.f;
        p.xt[bb * 32 + e] = f2bf(v);
      }
    }
    grid_barrier(p.cnt, p.gen, NBLK);
  }

  // ---------- ReLU(last) -> bf16 ----------
#pragma unroll
  for (int r = 0; r < 8; ++r) {
    int i = gtid + 8192 * r;
    float v = p.last[i];
    p.lastb[i] = f2bf(v > 0.f ? v : 0.f);
  }
  grid_barrier(p.cnt, p.gen, NBLK);

  // ---------- FF: y = relu(last) @ ff_w.T + ff_b  (256 waves, one 16x16 tile each)
  {
    v8f acc = zero8;
    int FM0 = (wg >> 4) * 16, FN0 = (wg & 15) * 16;
    for (int kk = 0; kk < 8; ++kk) {
      v16bf a = load_a(p.lastb, HIDN, FM0, kk * 32, lane);
      v16bf b = load_b(p.ff_wt, HIDN, kk * 32, FN0, lane);
      acc = wmma_bf16(a, b, acc);
    }
#pragma unroll
    for (int r = 0; r < 8; ++r) {
      int m = r + 8 * (lane >> 4), n = lane & 15;
      p.y[(FM0 + m) * HIDN + FN0 + n] = acc[r] + p.ff_b[FN0 + n];
    }
  }
  grid_barrier(p.cnt, p.gen, NBLK);

  // ---------- BatchNorm over batch: one wave per feature ----------
  {
    int j = wg;
    float vals[8];
    float s = 0.f, sq = 0.f;
#pragma unroll
    for (int r = 0; r < 8; ++r) {
      float v = p.y[(lane + 32 * r) * HIDN + j];
      vals[r] = v; s += v; sq += v * v;
    }
#pragma unroll
    for (int m = 16; m >= 1; m >>= 1) {
      s  += __shfl_xor(s,  m, 32);
      sq += __shfl_xor(sq, m, 32);
    }
    float mean  = s  * (1.f / BATCH);
    float var   = sq * (1.f / BATCH) - mean * mean;
    float scale = p.gamma[j] * rsqrtf(var + 1e-5f);
    float shift = p.beta[j] - mean * scale;
#pragma unroll
    for (int r = 0; r < 8; ++r)
      p.out[(lane + 32 * r) * HIDN + j] = vals[r] * scale + shift;
  }
}

extern "C" void kernel_launch(void* const* d_in, const int* in_sizes, int n_in,
                              void* d_out, int out_size, void* d_ws, size_t ws_size,
                              hipStream_t stream) {
  (void)in_sizes; (void)n_in; (void)out_size; (void)ws_size;
  size_t off = 0;
  auto take = [&](size_t bytes) -> void* {
    void* ptr = (void*)((char*)d_ws + off);
    off += (bytes + 255) & ~(size_t)255;
    return ptr;
  };

  Params P;
  P.tokens  = (const int*)d_in[0];
  P.lengths = (const int*)d_in[1];
  P.emb     = (const float*)d_in[2];
  P.w_ih0f  = (const float*)d_in[3];
  P.w_hh0f  = (const float*)d_in[4];
  P.b_ih0   = (const float*)d_in[5];
  P.b_hh0   = (const float*)d_in[6];
  P.w_ih1f  = (const float*)d_in[7];
  P.w_hh1f  = (const float*)d_in[8];
  P.b_ih1   = (const float*)d_in[9];
  P.b_hh1   = (const float*)d_in[10];
  P.ff_wf   = (const float*)d_in[11];
  P.ff_b    = (const float*)d_in[12];
  P.gamma   = (const float*)d_in[13];
  P.beta    = (const float*)d_in[14];

  P.cnt    = (int*)take(8);
  P.gen    = P.cnt + 1;
  P.w_hh0t = (unsigned short*)take(HIDN * G4 * 2);
  P.w_ih1t = (unsigned short*)take(HIDN * G4 * 2);
  P.w_hh1t = (unsigned short*)take(HIDN * G4 * 2);
  P.w_ih0t = (unsigned short*)take(32 * G4 * 2);
  P.ff_wt  = (unsigned short*)take(HIDN * HIDN * 2);
  P.h1b0   = (unsigned short*)take(BATCH * HIDN * 2);
  P.h1b1   = (unsigned short*)take(BATCH * HIDN * 2);
  P.h2b0   = (unsigned short*)take(BATCH * HIDN * 2);
  P.h2b1   = (unsigned short*)take(BATCH * HIDN * 2);
  P.xt     = (unsigned short*)take(BATCH * 32 * 2);
  P.lastb  = (unsigned short*)take(BATCH * HIDN * 2);
  P.c1     = (float*)take(BATCH * HIDN * 4);
  P.c2     = (float*)take(BATCH * HIDN * 4);
  P.last   = (float*)take(BATCH * HIDN * 4);
  P.y      = (float*)take(BATCH * HIDN * 4);
  P.out    = (float*)d_out;

  init_kernel<<<256, TPB, 0, stream>>>(P);
  lstm_persistent<<<NBLK, TPB, 0, stream>>>(P);
}
